// SelfAttention_2052994367667
// MI455X (gfx1250) — compile-verified, matrix operand order checked by reading, not run on hardware
//
#include <hip/hip_runtime.h>

#define BSZ   8
#define SEQ   2048
#define DM    1024
#define OUTN  1024
#define TOK   (BSZ * SEQ)   // 16384 rows for the projections

typedef __attribute__((ext_vector_type(16))) __bf16 v16bf;
typedef __attribute__((ext_vector_type(8)))  __bf16 v8bf;
typedef __attribute__((ext_vector_type(8)))  float  v8f;

__device__ __forceinline__ __bf16 cvt_bf16(float x)  { return (__bf16)x; }
__device__ __forceinline__ __bf16 cvt_bf16(__bf16 x) { return x; }

// ---------------------------------------------------------------------------
// Tiled GEMM: C[M,N] = A[M,K] * B[K,N] (+bias), bf16 WMMA, fp32 accumulate.
// Block tile 128x128, 8 waves (wave32), each wave computes 64x32 = 4x2
// subtiles of 16x16 via v_wmma_f32_16x16x32_bf16. K-step = 32.
// A,B staged in LDS as bf16 with padded rows (stride 40 -> 80B, 16B aligned).
// When A is already bf16 (the P·V GEMM), the A tile is staged with CDNA5
// async global->LDS copies (ASYNCcnt) instead of going through VGPRs.
// ---------------------------------------------------------------------------
template <typename TA, typename TB, typename TC, bool HAS_BIAS>
__global__ __launch_bounds__(256)
void gemm_wmma_kernel(const TA* __restrict__ A, const TB* __restrict__ Bm,
                      const float* __restrict__ bias, TC* __restrict__ C,
                      int M, int N, int K,
                      long long sA, long long sB, long long sC)
{
    __shared__ __bf16 As[128][40];   // [m][k], row stride 80B (16B aligned)
    __shared__ __bf16 Bs[128][40];   // [n][k] (B tile stored transposed)

    const int tid  = threadIdx.x;
    const int wid  = tid >> 5;
    const int lane = tid & 31;
    const int hh   = lane >> 4;      // which 16-lane half
    const int lp   = lane & 15;

    const int m0 = blockIdx.y * 128;
    const int n0 = blockIdx.x * 128;

    A  += (long long)blockIdx.z * sA;
    Bm += (long long)blockIdx.z * sB;
    C  += (long long)blockIdx.z * sC;

    const int wm = wid & 1;          // 2 wave-rows  * 64
    const int wn = wid >> 1;         // 4 wave-cols  * 32

    v8f acc[4][2];
#pragma unroll
    for (int i = 0; i < 4; ++i)
#pragma unroll
        for (int j = 0; j < 2; ++j)
#pragma unroll
            for (int e = 0; e < 8; ++e) acc[i][j][e] = 0.0f;

    for (int kb = 0; kb < K; kb += 32) {
        // ---- stage A tile: 128 rows x 32 k ----
        if constexpr (sizeof(TA) == 2) {
            // bf16 source, no conversion needed: use CDNA5 async copy
            // global -> LDS (GLOBAL_LOAD_ASYNC_TO_LDS_B128, ASYNCcnt).
            // 128 rows * 64B = 512 chunks of 16B; 256 lanes x 2 chunks.
            unsigned asbase = (unsigned)(unsigned long long)(&As[0][0]);
            const char* Abyte =
                (const char*)(A + (long long)m0 * K + kb);
#pragma unroll
            for (int i = 0; i < 2; ++i) {
                int idx = tid + i * 256;
                int m = idx >> 2, c = idx & 3;          // row, 16B chunk
                unsigned ldsoff = asbase + (unsigned)(m * 80 + c * 16);
                unsigned long long gaddr = (unsigned long long)
                    (Abyte + (long long)m * (K * 2) + c * 16);
                asm volatile("global_load_async_to_lds_b128 %0, %1, off"
                             :: "v"(ldsoff), "v"(gaddr) : "memory");
            }
        } else {
            // f32 source: load, convert to bf16 in VGPRs, ds-store.
#pragma unroll
            for (int i = 0; i < 16; ++i) {
                int idx = tid + i * 256;
                int m = idx >> 5, kk = idx & 31;
                As[m][kk] = cvt_bf16(A[(long long)(m0 + m) * K + kb + kk]);
            }
        }
        // ---- stage B tile transposed: Bs[n][k] = B[kb+k][n0+n] ----
#pragma unroll
        for (int i = 0; i < 16; ++i) {
            int idx = tid + i * 256;
            int n = idx & 127, kk = idx >> 7;
            Bs[n][kk] = cvt_bf16(Bm[(long long)(kb + kk) * N + n0 + n]);
        }
        if (kb + 32 < K)   // hint next A tile into cache (global_prefetch_b8)
            __builtin_prefetch(&A[(long long)(m0 + (tid >> 1)) * K + kb + 32], 0, 1);
        if constexpr (sizeof(TA) == 2)
            asm volatile("s_wait_asynccnt 0x0" ::: "memory");
        __syncthreads();

        // ---- per-lane fragments from LDS ----
        v16bf af[4], bfr[2];
#pragma unroll
        for (int sm = 0; sm < 4; ++sm) {
            int mr = wm * 64 + sm * 16 + lp;
            v8bf lo = *(const v8bf*)&As[mr][hh * 8];
            v8bf hi = *(const v8bf*)&As[mr][16 + hh * 8];
#pragma unroll
            for (int t = 0; t < 8; ++t) { af[sm][t] = lo[t]; af[sm][8 + t] = hi[t]; }
        }
#pragma unroll
        for (int sn = 0; sn < 2; ++sn) {
            int nr = wn * 32 + sn * 16 + lp;
            v8bf lo = *(const v8bf*)&Bs[nr][hh * 8];
            v8bf hi = *(const v8bf*)&Bs[nr][16 + hh * 8];
#pragma unroll
            for (int t = 0; t < 8; ++t) { bfr[sn][t] = lo[t]; bfr[sn][8 + t] = hi[t]; }
        }

#pragma unroll
        for (int sm = 0; sm < 4; ++sm)
#pragma unroll
            for (int sn = 0; sn < 2; ++sn)
                acc[sm][sn] = __builtin_amdgcn_wmma_f32_16x16x32_bf16(
                    false, af[sm], false, bfr[sn], (short)0, acc[sm][sn], false, false);

        __syncthreads();
    }

    // ---- epilogue: C layout = col lane%16, rows (lane/16)*8 + j ----
#pragma unroll
    for (int sm = 0; sm < 4; ++sm)
#pragma unroll
        for (int sn = 0; sn < 2; ++sn) {
            int col = n0 + wn * 32 + sn * 16 + lp;
            float bv = HAS_BIAS ? bias[col] : 0.0f;
#pragma unroll
            for (int j = 0; j < 8; ++j) {
                int row = m0 + wm * 64 + sm * 16 + hh * 8 + j;
                float val = acc[sm][sn][j] + bv;
                C[(long long)row * N + col] = (TC)val;
            }
        }
}

// ---------------------------------------------------------------------------
// Scores + softmax: one block per (batch, 16-query tile). 8 waves; wave w
// owns keys [w*256, w*256+256) as 16 subtiles of 16x16. Scores accumulate
// over D=1024 in 32-chunks (512 WMMA/wave), then scaled, masked, softmaxed
// (shfl_xor within 16-lane groups + cross-wave LDS reduce), and written as
// bf16 P to workspace. Q/K fragments stream from global bf16 (L2 resident).
// ---------------------------------------------------------------------------
__global__ __launch_bounds__(256)
void attn_scores_softmax_kernel(const __bf16* __restrict__ Qb,
                                const __bf16* __restrict__ Kb,
                                const int* __restrict__ mask,
                                __bf16* __restrict__ P)
{
    const int b   = blockIdx.y;
    const int q0  = blockIdx.x * 16;
    const int tid = threadIdx.x;
    const int wid = tid >> 5;
    const int lane = tid & 31;
    const int hh  = lane >> 4;
    const int lp  = lane & 15;

    const __bf16* Qrow  = Qb + ((long long)b * SEQ + q0 + lp) * DM;
    const __bf16* Kbase = Kb + (long long)b * SEQ * DM;

    v8f acc[16];
#pragma unroll
    for (int s = 0; s < 16; ++s)
#pragma unroll
        for (int e = 0; e < 8; ++e) acc[s][e] = 0.0f;

    for (int kc = 0; kc < DM; kc += 32) {
        v16bf af;
        {
            v8bf lo = *(const v8bf*)(Qrow + kc + hh * 8);
            v8bf hi = *(const v8bf*)(Qrow + kc + 16 + hh * 8);
#pragma unroll
            for (int t = 0; t < 8; ++t) { af[t] = lo[t]; af[8 + t] = hi[t]; }
        }
#pragma unroll
        for (int s = 0; s < 16; ++s) {
            const __bf16* Krow = Kbase + (long long)(wid * 256 + s * 16 + lp) * DM + kc;
            v8bf lo = *(const v8bf*)(Krow + hh * 8);
            v8bf hi = *(const v8bf*)(Krow + 16 + hh * 8);
            v16bf bfg;
#pragma unroll
            for (int t = 0; t < 8; ++t) { bfg[t] = lo[t]; bfg[8 + t] = hi[t]; }
            acc[s] = __builtin_amdgcn_wmma_f32_16x16x32_bf16(
                false, af, false, bfg, (short)0, acc[s], false, false);
        }
    }

    // scale + padding mask (mask==0 -> -inf)
    const float scale = 0.03125f;                 // 1/sqrt(1024)
    const float ninf  = -__builtin_inff();
#pragma unroll
    for (int s = 0; s < 16; ++s) {
        int key = wid * 256 + s * 16 + lp;
        bool valid = mask[b * SEQ + key] != 0;
#pragma unroll
        for (int j = 0; j < 8; ++j) {
            float v = acc[s][j] * scale;
            acc[s][j] = valid ? v : ninf;
        }
    }

    __shared__ float red[16][8];                  // [row][wave]

    // ---- row max ----
    float rmax[8];
#pragma unroll
    for (int j = 0; j < 8; ++j) {
        float m = acc[0][j];
#pragma unroll
        for (int s = 1; s < 16; ++s) m = fmaxf(m, acc[s][j]);
#pragma unroll
        for (int d = 1; d < 16; d <<= 1) m = fmaxf(m, __shfl_xor(m, d, 32));
        rmax[j] = m;
    }
    if (lp == 0) {
#pragma unroll
        for (int j = 0; j < 8; ++j) red[hh * 8 + j][wid] = rmax[j];
    }
    __syncthreads();
#pragma unroll
    for (int j = 0; j < 8; ++j) {
        float m = red[hh * 8 + j][0];
#pragma unroll
        for (int w = 1; w < 8; ++w) m = fmaxf(m, red[hh * 8 + j][w]);
        rmax[j] = m;
    }
    __syncthreads();

    // ---- exp + row sum ----
    float rsum[8];
#pragma unroll
    for (int j = 0; j < 8; ++j) rsum[j] = 0.0f;
#pragma unroll
    for (int s = 0; s < 16; ++s)
#pragma unroll
        for (int j = 0; j < 8; ++j) {
            float e = __expf(acc[s][j] - rmax[j]);
            acc[s][j] = e;
            rsum[j] += e;
        }
#pragma unroll
    for (int j = 0; j < 8; ++j)
#pragma unroll
        for (int d = 1; d < 16; d <<= 1) rsum[j] += __shfl_xor(rsum[j], d, 32);
    if (lp == 0) {
#pragma unroll
        for (int j = 0; j < 8; ++j) red[hh * 8 + j][wid] = rsum[j];
    }
    __syncthreads();
#pragma unroll
    for (int j = 0; j < 8; ++j) {
        float t = 0.0f;
#pragma unroll
        for (int w = 0; w < 8; ++w) t += red[hh * 8 + j][w];
        rsum[j] = t;
    }

    // ---- normalize + store bf16 weights ----
    __bf16* Pb = P + (long long)b * SEQ * SEQ;
#pragma unroll
    for (int j = 0; j < 8; ++j) {
        float inv = 1.0f / rsum[j];
        int row = q0 + hh * 8 + j;
#pragma unroll
        for (int s = 0; s < 16; ++s) {
            int key = wid * 256 + s * 16 + lp;
            Pb[(long long)row * SEQ + key] = (__bf16)(acc[s][j] * inv);
        }
    }
}

// ---------------------------------------------------------------------------
extern "C" void kernel_launch(void* const* d_in, const int* in_sizes, int n_in,
                              void* d_out, int out_size, void* d_ws, size_t ws_size,
                              hipStream_t stream) {
    const float* q   = (const float*)d_in[0];
    const float* k   = (const float*)d_in[1];
    const float* v   = (const float*)d_in[2];
    const float* Wq  = (const float*)d_in[3];
    const float* bq  = (const float*)d_in[4];
    const float* Wk  = (const float*)d_in[5];
    const float* bk  = (const float*)d_in[6];
    const float* Wv  = (const float*)d_in[7];
    const float* bv  = (const float*)d_in[8];
    const int*  mask = (const int*) d_in[9];
    float* out = (float*)d_out;

    // Workspace: Qb/Kb/Vb bf16 (32 MB each) + P bf16 (64 MB) = 160 MB
    __bf16* Qb = (__bf16*)d_ws;
    __bf16* Kb = Qb + (size_t)TOK * DM;
    __bf16* Vb = Kb + (size_t)TOK * DM;
    __bf16* Pw = Vb + (size_t)TOK * DM;

    dim3 blk(256);

    // Projections: [16384,1024] x [1024,1024] + bias -> bf16
    dim3 gproj(OUTN / 128, TOK / 128, 1);
    gemm_wmma_kernel<float, float, __bf16, true><<<gproj, blk, 0, stream>>>(
        q, Wq, bq, Qb, TOK, OUTN, DM, 0, 0, 0);
    gemm_wmma_kernel<float, float, __bf16, true><<<gproj, blk, 0, stream>>>(
        k, Wk, bk, Kb, TOK, OUTN, DM, 0, 0, 0);
    gemm_wmma_kernel<float, float, __bf16, true><<<gproj, blk, 0, stream>>>(
        v, Wv, bv, Vb, TOK, OUTN, DM, 0, 0, 0);

    // Scores + softmax -> bf16 attention weights P [8,2048,2048]
    attn_scores_softmax_kernel<<<dim3(SEQ / 16, BSZ), blk, 0, stream>>>(
        Qb, Kb, mask, Pw);

    // Output: per-batch P[2048,2048] x V[2048,1024] -> f32 out
    gemm_wmma_kernel<__bf16, __bf16, float, false><<<dim3(OUTN / 128, SEQ / 128, BSZ), blk, 0, stream>>>(
        Pw, Vb, nullptr, out, SEQ, OUTN, SEQ,
        (long long)SEQ * SEQ, (long long)SEQ * DM, (long long)SEQ * DM);
}
